// STEM_Layer_82935818486499
// MI455X (gfx1250) — matrix operand order repaired and checked by reading, not compiled, not run on hardware
//
#include <hip/hip_runtime.h>
#include <math.h>

typedef __bf16 bf16;
typedef __attribute__((ext_vector_type(16))) __bf16 v16bf;
typedef __attribute__((ext_vector_type(8)))  __bf16 v8bf;
typedef __attribute__((ext_vector_type(2)))  __bf16 v2bf;
typedef __attribute__((ext_vector_type(8)))  float  v8f;

// Problem constants
constexpr int Tt = 3, Ee = 4, Ss = 4;
constexpr int Dd = 1024, H1d = 1024, H2d = 512, Gg = 512, Oo = 16;
constexpr int Bb = 8192;
constexpr int NE = Tt * Ee;          // 12 experts
constexpr int NOUT = NE + Ss;        // 16 mixture inputs

// ---------------- workspace layout (bytes) ----------------
constexpr size_t XB_OFF    = 0;                                        // bf16 xs        (4*B*D)
constexpr size_t XB_SZ     = (size_t)(Tt + 1) * Bb * Dd * 2;
constexpr size_t XG_OFF    = XB_OFF + XB_SZ;                           // bf16 xg        (4*B*D)
constexpr size_t XG_SZ     = XB_SZ;
constexpr size_t WSW1_OFF  = XG_OFF + XG_SZ;                           // bf16 sw1
constexpr size_t WSW1_SZ   = (size_t)NE * Dd * H1d * 2;
constexpr size_t WSW2_OFF  = WSW1_OFF + WSW1_SZ;                       // bf16 sw2
constexpr size_t WSW2_SZ   = (size_t)NE * H1d * H2d * 2;
constexpr size_t WSHW1_OFF = WSW2_OFF + WSW2_SZ;                       // bf16 shw1
constexpr size_t WSHW1_SZ  = (size_t)Ss * Dd * H1d * 2;
constexpr size_t WSHW2_OFF = WSHW1_OFF + WSHW1_SZ;                     // bf16 shw2
constexpr size_t WSHW2_SZ  = (size_t)Ss * H1d * H2d * 2;
constexpr size_t WGW1_OFF  = WSHW2_OFF + WSHW2_SZ;                     // bf16 gw1
constexpr size_t WGW1_SZ   = (size_t)(Tt + 1) * Dd * Gg * 2;
constexpr size_t WGW2_OFF  = WGW1_OFF + WGW1_SZ;                       // bf16 gw2
constexpr size_t WGW2_SZ   = (size_t)(Tt + 1) * Gg * Oo * 2;
constexpr size_t HBUF_OFF  = WGW2_OFF + WGW2_SZ;                       // bf16 hidden (reused: h / hs / gh)
constexpr size_t HBUF_SZ   = (size_t)NE * Bb * H1d * 2;
constexpr size_t AOUT_OFF  = HBUF_OFF + HBUF_SZ;                       // bf16 all_out (16*B*512)
constexpr size_t AOUT_SZ   = (size_t)NOUT * Bb * H2d * 2;
constexpr size_t LOGIT_OFF = AOUT_OFF + AOUT_SZ;                       // f32 logits (4*B*16)
constexpr size_t LOGIT_SZ  = (size_t)(Tt + 1) * Bb * Oo * 4;
constexpr size_t GATE_OFF  = LOGIT_OFF + LOGIT_SZ;                     // f32 gate (4*B*16)

// ---------------- conversion kernels ----------------
__global__ void cvt_x_kernel(const float* __restrict__ xs, bf16* __restrict__ xb,
                             bf16* __restrict__ xg, int per, int total) {
    for (int idx = blockIdx.x * blockDim.x + threadIdx.x; idx < total;
         idx += gridDim.x * blockDim.x) {
        float v = xs[idx];
        xb[idx] = (bf16)v;
        int i = idx / per;
        int rem = idx - i * per;
        float g = (i < Tt) ? v + xs[(size_t)Tt * per + rem] : v;
        xg[idx] = (bf16)g;
    }
}

__global__ void cvt_w_kernel(const float* __restrict__ w, bf16* __restrict__ o, int n) {
    for (int idx = blockIdx.x * blockDim.x + threadIdx.x; idx < n;
         idx += gridDim.x * blockDim.x)
        o[idx] = (bf16)w[idx];
}

// ---------------- fused GEMM + bias + (ReLU) ----------------
// C[z] = act(A[z/aDiv] * W[z] + bias[z]),  A: MxK row-major, W: KxN row-major
template <bool RELU, typename OUT_T>
__global__ __launch_bounds__(256) void gemm_bias_act(
    const bf16* __restrict__ A, const bf16* __restrict__ W,
    const float* __restrict__ bias, OUT_T* __restrict__ C,
    int M, int N, int K,
    long long strideA, int aDiv, long long strideW, int strideBias, long long strideC) {
    constexpr int BM = 128, BN = 128, BK = 32;
    __shared__ __align__(64) bf16 As[2][BM][BK];   // row-major  [m][k]
    __shared__ __align__(64) bf16 Bs[2][BN][BK];   // col-major  [n][k]

    const int tid = threadIdx.x;
    const int z   = blockIdx.z;
    const int m0  = blockIdx.y * BM;
    const int n0  = blockIdx.x * BN;

    const bf16* Az = A + (size_t)(z / aDiv) * strideA;
    const bf16* Wz = W + (size_t)z * strideW;
    const float* bz = bias + (size_t)z * strideBias;
    OUT_T* Cz = C + (size_t)z * strideC;

    const int wave = tid >> 5;
    const int lane = tid & 31;
    const int l15  = lane & 15;
    const int hi   = lane >> 4;
    const int mw   = (wave & 3) * 32;   // wave row offset in tile
    const int nw   = (wave >> 2) * 64;  // wave col offset in tile

    // A tile: 512 16B chunks, 2 per thread  (async DMA global -> LDS)
    const int c0  = tid * 2;
    const int ar0 = c0 >> 2,        ak0 = (c0 & 3) * 8;
    const int ar1 = (c0 + 1) >> 2,  ak1 = ((c0 + 1) & 3) * 8;
    // B tile: one 2(K) x 8(N) sub-block per thread, packed-pair transpose
    const int bkp = (tid >> 4) * 2;        // K row pair: 0,2,...,30
    const int bnc = (tid & 15) * 8;        // N chunk: 0..120

    v8f acc[2][4] = {};

    auto load_tile = [&](int kt, int buf) {
        // ---- A: async DMA straight into LDS (ASYNCcnt), no VGPR round-trip ----
        {
            const bf16* ga0 = Az + (size_t)(m0 + ar0) * K + (size_t)kt * BK + ak0;
            const bf16* ga1 = Az + (size_t)(m0 + ar1) * K + (size_t)kt * BK + ak1;
            unsigned la0 = (unsigned)(uintptr_t)&As[buf][ar0][ak0];
            unsigned la1 = (unsigned)(uintptr_t)&As[buf][ar1][ak1];
            asm volatile("global_load_async_to_lds_b128 %0, %1, off"
                         :: "v"(la0), "v"(ga0) : "memory");
            asm volatile("global_load_async_to_lds_b128 %0, %1, off"
                         :: "v"(la1), "v"(ga1) : "memory");
        }
        // ---- B: load 2 K-rows of an 8-wide N chunk, store K-pairs (b32) col-major ----
        v8bf b0 = *(const v8bf*)(Wz + (size_t)(kt * BK + bkp) * N + n0 + bnc);
        v8bf b1 = *(const v8bf*)(Wz + (size_t)(kt * BK + bkp + 1) * N + n0 + bnc);
#pragma unroll
        for (int j = 0; j < 8; ++j) {
            v2bf p;
            p[0] = b0[j];
            p[1] = b1[j];
            *(v2bf*)&Bs[buf][bnc + j][bkp] = p;
        }
    };

    const int KT = K / BK;
    load_tile(0, 0);
    int buf = 0;
    for (int kt = 0; kt < KT; ++kt) {
        asm volatile("s_wait_asynccnt 0" ::: "memory");  // drain async A-tile DMA
        __syncthreads();
        if (kt + 1 < KT) load_tile(kt + 1, buf ^ 1);     // overlaps compute below

        // A fragments: lane<16 -> row l15, K {0..15}; lane>=16 -> K {16..31}
        union { v16bf v; v8bf h[2]; } af[2];
#pragma unroll
        for (int mi = 0; mi < 2; ++mi) {
            const int row = mw + mi * 16 + l15;
            af[mi].h[0] = *(const v8bf*)&As[buf][row][hi * 16];
            af[mi].h[1] = *(const v8bf*)&As[buf][row][hi * 16 + 8];
        }
#pragma unroll
        for (int ni = 0; ni < 4; ++ni) {
            const int col = nw + ni * 16 + l15;
            // B fragment: lane = column, 16 contiguous K values (hi selects K 0..15 / 16..31)
            v16bf bfr = *(const v16bf*)&Bs[buf][col][hi * 16];
#pragma unroll
            for (int mi = 0; mi < 2; ++mi)
                acc[mi][ni] = __builtin_amdgcn_wmma_f32_16x16x32_bf16(
                    false, af[mi].v, false, bfr, (short)0, acc[mi][ni], false, false);
        }
        buf ^= 1;
    }

    // epilogue: C VGPR r -> (M = r + 8*hi, N = l15)
#pragma unroll
    for (int ni = 0; ni < 4; ++ni) {
        const int col = n0 + nw + ni * 16 + l15;
        const float bcol = bz[col];
#pragma unroll
        for (int mi = 0; mi < 2; ++mi) {
            const int rbase = m0 + mw + mi * 16 + hi * 8;
#pragma unroll
            for (int r = 0; r < 8; ++r) {
                float v = acc[mi][ni][r] + bcol;
                if (RELU) v = fmaxf(v, 0.0f);
                Cz[(size_t)(rbase + r) * N + col] = (OUT_T)v;
            }
        }
    }
}

// ---------------- gate logits: [B x 512] x [512 x 16] per i ----------------
__global__ __launch_bounds__(256) void gate_logits_kernel(
    const bf16* __restrict__ gh, const bf16* __restrict__ w2,
    const float* __restrict__ b2, float* __restrict__ logits) {
    __shared__ __align__(64) bf16 Wl[Oo][Gg];   // col-major weight slice
    const int tid = threadIdx.x;
    const int z = blockIdx.y;
    const int rowBase = blockIdx.x * 128;

    const bf16* wz = w2 + (size_t)z * Gg * Oo;
#pragma unroll
    for (int j = 0; j < 32; ++j) {
        int e = j * 256 + tid;          // e = k*16 + n
        Wl[e & 15][e >> 4] = wz[e];
    }
    __syncthreads();

    const int wave = tid >> 5, lane = tid & 31, l15 = lane & 15, hi = lane >> 4;
    const int row = rowBase + wave * 16 + l15;
    const bf16* az = gh + ((size_t)z * Bb + row) * Gg;

    v8f acc = {};
    for (int k0 = 0; k0 < Gg; k0 += 32) {
        union { v16bf v; v8bf h[2]; } af;
        af.h[0] = *(const v8bf*)(az + k0 + hi * 16);
        af.h[1] = *(const v8bf*)(az + k0 + hi * 16 + 8);
        v16bf bfr = *(const v16bf*)&Wl[l15][k0 + hi * 16];
        acc = __builtin_amdgcn_wmma_f32_16x16x32_bf16(false, af.v, false, bfr,
                                                      (short)0, acc, false, false);
    }
    const float bcol = b2[z * Oo + l15];
    const int rb = rowBase + wave * 16 + hi * 8;
#pragma unroll
    for (int r = 0; r < 8; ++r)
        logits[((size_t)z * Bb + rb + r) * Oo + l15] = acc[r] + bcol;
}

// ---------------- softmax over 16 logits per row ----------------
__global__ void softmax16_kernel(const float* __restrict__ logits,
                                 float* __restrict__ gate, int rows) {
    int r = blockIdx.x * blockDim.x + threadIdx.x;
    if (r >= rows) return;
    const float* in = logits + (size_t)r * Oo;
    float m = in[0];
#pragma unroll
    for (int i = 1; i < Oo; ++i) m = fmaxf(m, in[i]);
    float e[Oo], s = 0.0f;
#pragma unroll
    for (int i = 0; i < Oo; ++i) { e[i] = __expf(in[i] - m); s += e[i]; }
    const float inv = 1.0f / s;
    float* o = gate + (size_t)r * Oo;
#pragma unroll
    for (int i = 0; i < Oo; ++i) o[i] = e[i] * inv;
}

// ---------------- mixture: outs[i,b,:] = sum_n gate[i,b,n] * all_out[n,b,:] ----------------
__global__ __launch_bounds__(256) void combine_kernel(
    const bf16* __restrict__ allout, const float* __restrict__ gate,
    float* __restrict__ out) {
    __shared__ bf16 Ao[NOUT][H2d];
    __shared__ float Gs[Tt + 1][Oo];
    const int b = blockIdx.x, tid = threadIdx.x;
#pragma unroll
    for (int j = 0; j < 32; ++j) {
        int e = j * 256 + tid;          // e = n*512 + h
        int n = e >> 9, h = e & 511;
        Ao[n][h] = allout[((size_t)n * Bb + b) * H2d + h];
    }
    if (tid < (Tt + 1) * Oo)
        Gs[tid >> 4][tid & 15] = gate[((size_t)(tid >> 4) * Bb + b) * Oo + (tid & 15)];
    __syncthreads();
#pragma unroll
    for (int i = 0; i < Tt + 1; ++i) {
        for (int h = tid; h < H2d; h += 256) {
            float a = 0.0f;
#pragma unroll
            for (int n = 0; n < NOUT; ++n) a += Gs[i][n] * (float)Ao[n][h];
            out[((size_t)i * Bb + b) * H2d + h] = a;
        }
    }
}

// ---------------- launcher ----------------
extern "C" void kernel_launch(void* const* d_in, const int* in_sizes, int n_in,
                              void* d_out, int out_size, void* d_ws, size_t ws_size,
                              hipStream_t stream) {
    const float* xs   = (const float*)d_in[0];
    const float* sw1  = (const float*)d_in[1];
    const float* sb1  = (const float*)d_in[2];
    const float* sw2  = (const float*)d_in[3];
    const float* sb2  = (const float*)d_in[4];
    const float* shw1 = (const float*)d_in[5];
    const float* shb1 = (const float*)d_in[6];
    const float* shw2 = (const float*)d_in[7];
    const float* shb2 = (const float*)d_in[8];
    const float* gw1  = (const float*)d_in[9];
    const float* gb1  = (const float*)d_in[10];
    const float* gw2  = (const float*)d_in[11];
    const float* gb2  = (const float*)d_in[12];

    char* ws = (char*)d_ws;
    bf16*  xb     = (bf16*)(ws + XB_OFF);
    bf16*  xg     = (bf16*)(ws + XG_OFF);
    bf16*  wsw1   = (bf16*)(ws + WSW1_OFF);
    bf16*  wsw2   = (bf16*)(ws + WSW2_OFF);
    bf16*  wshw1  = (bf16*)(ws + WSHW1_OFF);
    bf16*  wshw2  = (bf16*)(ws + WSHW2_OFF);
    bf16*  wgw1   = (bf16*)(ws + WGW1_OFF);
    bf16*  wgw2   = (bf16*)(ws + WGW2_OFF);
    bf16*  hbuf   = (bf16*)(ws + HBUF_OFF);
    bf16*  allout = (bf16*)(ws + AOUT_OFF);
    float* logits = (float*)(ws + LOGIT_OFF);
    float* gate   = (float*)(ws + GATE_OFF);
    float* out    = (float*)d_out;

    // 1) precision conversion (bandwidth-bound, one pass)
    cvt_x_kernel<<<4096, 256, 0, stream>>>(xs, xb, xg, Bb * Dd, (Tt + 1) * Bb * Dd);
    cvt_w_kernel<<<2048, 256, 0, stream>>>(sw1,  wsw1,  NE * Dd * H1d);
    cvt_w_kernel<<<2048, 256, 0, stream>>>(sw2,  wsw2,  NE * H1d * H2d);
    cvt_w_kernel<<<1024, 256, 0, stream>>>(shw1, wshw1, Ss * Dd * H1d);
    cvt_w_kernel<<<1024, 256, 0, stream>>>(shw2, wshw2, Ss * H1d * H2d);
    cvt_w_kernel<<<1024, 256, 0, stream>>>(gw1,  wgw1,  (Tt + 1) * Dd * Gg);
    cvt_w_kernel<<<64,   256, 0, stream>>>(gw2,  wgw2,  (Tt + 1) * Gg * Oo);

    // 2) expert MLPs (z = t*E + e)
    gemm_bias_act<true, bf16><<<dim3(H1d / 128, Bb / 128, NE), 256, 0, stream>>>(
        xb, wsw1, sb1, hbuf, Bb, H1d, Dd,
        (long long)Bb * Dd, Ee, (long long)Dd * H1d, H1d, (long long)Bb * H1d);
    gemm_bias_act<true, bf16><<<dim3(H2d / 128, Bb / 128, NE), 256, 0, stream>>>(
        hbuf, wsw2, sb2, allout, Bb, H2d, H1d,
        (long long)Bb * H1d, 1, (long long)H1d * H2d, H2d, (long long)Bb * H2d);

    // 3) shared MLPs (A = xsh for every z), hidden buffer reused
    gemm_bias_act<true, bf16><<<dim3(H1d / 128, Bb / 128, Ss), 256, 0, stream>>>(
        xb + (size_t)Tt * Bb * Dd, wshw1, shb1, hbuf, Bb, H1d, Dd,
        0LL, 1, (long long)Dd * H1d, H1d, (long long)Bb * H1d);
    gemm_bias_act<true, bf16><<<dim3(H2d / 128, Bb / 128, Ss), 256, 0, stream>>>(
        hbuf, wshw2, shb2, allout + (size_t)NE * Bb * H2d, Bb, H2d, H1d,
        (long long)Bb * H1d, 1, (long long)H1d * H2d, H2d, (long long)Bb * H2d);

    // 4) gate hidden (hbuf reused as gh), logits, softmax
    gemm_bias_act<true, bf16><<<dim3(Gg / 128, Bb / 128, Tt + 1), 256, 0, stream>>>(
        xg, wgw1, gb1, hbuf, Bb, Gg, Dd,
        (long long)Bb * Dd, 1, (long long)Dd * Gg, Gg, (long long)Bb * Gg);
    gate_logits_kernel<<<dim3(Bb / 128, Tt + 1), 256, 0, stream>>>(hbuf, wgw2, gb2, logits);
    softmax16_kernel<<<((Tt + 1) * Bb + 255) / 256, 256, 0, stream>>>(
        logits, gate, (Tt + 1) * Bb);

    // 5) mixture
    combine_kernel<<<Bb, 256, 0, stream>>>(allout, gate, out);
}
